// GeminiFusionAttention_27908697489630
// MI455X (gfx1250) — compile-verified, matrix operand order checked by reading, not run on hardware
//
#include <hip/hip_runtime.h>
#include <math.h>

typedef __bf16 bf16_t;
typedef bf16_t v8bf  __attribute__((ext_vector_type(8)));
typedef bf16_t v16bf __attribute__((ext_vector_type(16)));
typedef float  v8f   __attribute__((ext_vector_type(8)));
typedef float  v4f   __attribute__((ext_vector_type(4)));

#define TILE_M 128
#define TILE_N 128
#define TILE_K 32
// LDS tile row stride: 32 bf16 = 64 bytes; tile = 128*64 = 8192 bytes

__device__ inline bf16_t f2bf(float f) {
  unsigned int u = __builtin_bit_cast(unsigned int, f);
  unsigned int r = u + 0x7FFFu + ((u >> 16) & 1u);   // round-to-nearest-even
  unsigned short s = (unsigned short)(r >> 16);
  return __builtin_bit_cast(bf16_t, s);
}

// CDNA5 async global->LDS copy, 16 bytes per lane, tracked by ASYNCcnt.
__device__ inline void async_copy_b128(unsigned lds_off, const void* gaddr) {
  asm volatile("global_load_async_to_lds_b128 %0, %1, off"
               :: "v"(lds_off), "v"(gaddr) : "memory");
}
#define ASYNC_WAIT(N) asm volatile("s_wait_asynccnt " #N ::: "memory")

// ---------------------------------------------------------------------------
// C[M,N] = A[M,K](bf16) @ BT[N,K](bf16)^T  (+C if accumulate) (+bias[N] if set)
// Double-buffered async-LDS pipeline + v_wmma_f32_16x16x32_bf16.
// ---------------------------------------------------------------------------
__global__ __launch_bounds__(256)
void gemm_bf16_wmma_async(const bf16_t* __restrict__ A, const bf16_t* __restrict__ BT,
                          float* __restrict__ C, const float* __restrict__ bias,
                          int M, int K, int N, int accumulate)
{
  __shared__ bf16_t lds[2 * 2 * TILE_M * TILE_K];   // [buf][A|B][128][32] = 32KB
  const unsigned ldsbase = (unsigned)(unsigned long long)(&lds[0]);

  const int tid  = threadIdx.x;
  const int lane = tid & 31;
  const int wave = tid >> 5;
  const int m0 = blockIdx.y * TILE_M;
  const int n0 = blockIdx.x * TILE_N;
  const int wm = (wave & 3) * 32;    // 4 waves along M
  const int wn = (wave >> 2) * 64;   // 2 waves along N

  v8f acc[2][4];
  #pragma unroll
  for (int i = 0; i < 2; i++)
    #pragma unroll
    for (int j = 0; j < 4; j++)
      #pragma unroll
      for (int r = 0; r < 8; r++) acc[i][j][r] = 0.0f;

  // 512 16-byte chunks per 8KB tile; each thread owns chunks {tid, tid+256}.
  // chunk>>2 = row (A) / col (B), chunk&3 = 16B sub-chunk within the 64B row.
  int crow[2], cko[2], bcol_g[2];
  #pragma unroll
  for (int p = 0; p < 2; p++) {
    int chunk = tid + p * 256;
    crow[p] = chunk >> 2;
    cko[p]  = (chunk & 3) * 8;                 // bf16-element offset in row
    int bc  = n0 + crow[p];
    bcol_g[p] = (bc < N) ? bc : (N - 1);       // clamp: garbage cols never stored
  }

  const int nkt = K / TILE_K;
  auto issue_tile = [&](int kt, int buf) {
    unsigned aoff = ldsbase + (unsigned)buf * 16384u;
    unsigned boff = aoff + 8192u;
    #pragma unroll
    for (int p = 0; p < 2; p++) {
      async_copy_b128(aoff + (unsigned)(crow[p] * 64 + cko[p] * 2),
                      A + (size_t)(m0 + crow[p]) * K + kt * TILE_K + cko[p]);
      async_copy_b128(boff + (unsigned)(crow[p] * 64 + cko[p] * 2),
                      BT + (size_t)bcol_g[p] * K + kt * TILE_K + cko[p]);
    }
  };

  issue_tile(0, 0);   // prologue: tile 0 in flight (4 async ops per wave)

  const int halfSel = (lane >> 4) & 1;   // ISA 7.12.2 16-bit A/B fragment layout
  const int lmod    = lane & 15;

  for (int kt = 0; kt < nkt; kt++) {
    const int cur = kt & 1;
    if (kt + 1 < nkt) { issue_tile(kt + 1, cur ^ 1); ASYNC_WAIT(4); }
    else              { ASYNC_WAIT(0); }
    __syncthreads();   // all waves' copies for tile kt are resident

    const bf16_t* bufA = lds + cur * 8192;   // elements (16384 B / 2)
    const bf16_t* bufB = bufA + 4096;        // +8192 B

    v16bf afrag[2], bfrag[4];
    #pragma unroll
    for (int i = 0; i < 2; i++) {
      const bf16_t* base = bufA + (wm + i * 16 + lmod) * TILE_K;
      v8bf lo = *reinterpret_cast<const v8bf*>(base + halfSel * 8);
      v8bf hi = *reinterpret_cast<const v8bf*>(base + 16 + halfSel * 8);
      afrag[i] = __builtin_shufflevector(lo, hi, 0,1,2,3,4,5,6,7,8,9,10,11,12,13,14,15);
    }
    #pragma unroll
    for (int j = 0; j < 4; j++) {
      const bf16_t* base = bufB + (wn + j * 16 + lmod) * TILE_K;
      v8bf lo = *reinterpret_cast<const v8bf*>(base + halfSel * 8);
      v8bf hi = *reinterpret_cast<const v8bf*>(base + 16 + halfSel * 8);
      bfrag[j] = __builtin_shufflevector(lo, hi, 0,1,2,3,4,5,6,7,8,9,10,11,12,13,14,15);
    }

    #pragma unroll
    for (int i = 0; i < 2; i++)
      #pragma unroll
      for (int j = 0; j < 4; j++)
        acc[i][j] = __builtin_amdgcn_wmma_f32_16x16x32_bf16(
            false, afrag[i], false, bfrag[j], (short)0, acc[i][j], false, false);

    __syncthreads();   // safe to overwrite buf cur next-next iteration
  }

  // store (f32 D layout: VGPR r -> row r / 8+r, lane%16 -> column)
  const int rowAdd = halfSel * 8;
  #pragma unroll
  for (int i = 0; i < 2; i++) {
    #pragma unroll
    for (int j = 0; j < 4; j++) {
      int gcol = n0 + wn + j * 16 + lmod;
      if (gcol >= N) continue;
      #pragma unroll
      for (int r = 0; r < 8; r++) {
        int grow = m0 + wm + i * 16 + rowAdd + r;
        size_t idx = (size_t)grow * N + gcol;
        float v = acc[i][j][r];
        if (accumulate) v += C[idx];
        if (bias) v += bias[gcol];
        C[idx] = v;
      }
    }
  }
}

// ---------------------------------------------------------------------------
// f32 -> bf16 bulk conversion (8 elems/thread, vectorized)
// ---------------------------------------------------------------------------
__global__ __launch_bounds__(256)
void f32_to_bf16_kernel(const float* __restrict__ in, bf16_t* __restrict__ out, size_t n)
{
  size_t i = ((size_t)blockIdx.x * 256 + threadIdx.x) * 8;
  if (i >= n) return;
  v4f a = *reinterpret_cast<const v4f*>(in + i);
  v4f b = *reinterpret_cast<const v4f*>(in + i + 4);
  v8bf o;
  o[0]=f2bf(a.x); o[1]=f2bf(a.y); o[2]=f2bf(a.z); o[3]=f2bf(a.w);
  o[4]=f2bf(b.x); o[5]=f2bf(b.y); o[6]=f2bf(b.z); o[7]=f2bf(b.w);
  *reinterpret_cast<v8bf*>(out + i) = o;
}

// W[K,N] f32  ->  WT[N,K] bf16   (weights are tiny; layout for async GEMM tiles)
__global__ __launch_bounds__(256)
void convert_weight_T(const float* __restrict__ W, bf16_t* __restrict__ WT, int K, int N)
{
  int k = blockIdx.x * 32 + (threadIdx.x & 31);
  int n = blockIdx.y * 8 + (threadIdx.x >> 5);
  if (k < K && n < N)
    WT[(size_t)n * K + k] = f2bf(W[(size_t)k * N + n]);
}

// nk[j] = sum_k noise[k] * Wk[k,j]
__global__ void noise_k_kernel(const float* __restrict__ noise,
                               const float* __restrict__ Wk,
                               float* __restrict__ nk, int D)
{
  int j = blockIdx.x * 256 + threadIdx.x;
  if (j >= D) return;
  float s = 0.f;
  for (int k = 0; k < D; k++) s += noise[k] * Wk[(size_t)k * D + j];
  nk[j] = s;
}

// One wave per row: gate MLP second layer + 2-key softmax + fused output.
__global__ __launch_bounds__(256)
void fuse_epilogue(const float* __restrict__ xs, const float* __restrict__ q,
                   const float* __restrict__ kself, const float* __restrict__ nk,
                   const float* __restrict__ kcross,
                   const float* __restrict__ vself, const float* __restrict__ vcross,
                   const float* __restrict__ H, const float* __restrict__ W2,
                   const float* __restrict__ b2, float* __restrict__ out, int rows)
{
  const int lane = threadIdx.x & 31;
  const int warp = threadIdx.x >> 5;
  const int row = blockIdx.x * 8 + warp;
  if (row >= rows) return;
  const size_t base  = (size_t)row * 768;
  const size_t hbase = (size_t)row * 192;

  float ds = 0.f, dc = 0.f, p = 0.f;
  #pragma unroll
  for (int e = 0; e < 24; e++) {
    int idx = e * 32 + lane;
    float qv = q[base + idx];
    ds += qv * (kself[base + idx] + nk[idx]);
    dc += qv * kcross[base + idx];
  }
  #pragma unroll
  for (int e = 0; e < 6; e++) {
    int j = e * 32 + lane;
    float hv = H[hbase + j];
    float g = 0.5f * hv * (1.f + erff(hv * 0.7071067811865475f));  // exact gelu
    p += g * W2[j];
  }
  for (int off = 16; off > 0; off >>= 1) {
    ds += __shfl_xor(ds, off, 32);
    dc += __shfl_xor(dc, off, 32);
    p  += __shfl_xor(p,  off, 32);
  }
  const float scale = 0.03608439182435161f;  // 768^-0.5
  float rel = 1.f / (1.f + expf(-(p + b2[0])));
  float a = ds * scale;
  float b = dc * rel * scale;
  float m = fmaxf(a, b);
  float e0 = expf(a - m), e1 = expf(b - m);
  float inv = 1.f / (e0 + e1);
  float w0 = e0 * inv, w1 = e1 * inv;
  #pragma unroll
  for (int e = 0; e < 24; e++) {
    int idx = e * 32 + lane;
    out[base + idx] = xs[base + idx] + w0 * vself[base + idx] + w1 * vcross[base + idx];
  }
}

extern "C" void kernel_launch(void* const* d_in, const int* in_sizes, int n_in,
                              void* d_out, int out_size, void* d_ws, size_t ws_size,
                              hipStream_t stream)
{
  (void)in_sizes; (void)n_in; (void)out_size; (void)ws_size;
  const float* x1    = (const float*)d_in[0];
  const float* x2    = (const float*)d_in[1];
  const float* Wq    = (const float*)d_in[2];
  const float* Wk    = (const float*)d_in[3];
  const float* Wv    = (const float*)d_in[4];
  const float* noise = (const float*)d_in[5];
  const float* W1    = (const float*)d_in[6];
  const float* b1    = (const float*)d_in[7];
  const float* W2    = (const float*)d_in[8];
  const float* b2    = (const float*)d_in[9];
  float* out = (float*)d_out;

  const int M = 32 * 1024, D = 768, Hh = 192;
  const size_t S  = (size_t)M * D;       // 25,165,824
  const size_t Sh = (size_t)M * Hh;      //  6,291,456
  const size_t Dw = (size_t)D * D;       //    589,824
  const size_t Dh = (size_t)D * Hh;      //    147,456

  // f32 workspace region
  float* ws = (float*)d_ws;
  float *Q1 = ws,         *Q2 = ws + S,
        *K1 = ws + 2 * S, *K2 = ws + 3 * S,
        *V1 = ws + 4 * S, *V2 = ws + 5 * S,
        *H1 = ws + 6 * S, *H2 = ws + 6 * S + Sh,
        *NK = ws + 6 * S + 2 * Sh;
  // bf16 workspace region (16B-aligned: offset is a multiple of 4 floats)
  bf16_t* bws = (bf16_t*)(ws + 6 * S + 2 * Sh + 768);
  bf16_t *x1b  = bws,             *x2b  = bws + S,
         *WqT  = bws + 2 * S,     *WkT  = WqT + Dw,  *WvT = WkT + Dw,
         *W1aT = WvT + Dw,        *W1bT = W1aT + Dh;

  dim3 blk(256);

  // 1) one-time bf16 conversions (activations row-major, weights transposed)
  f32_to_bf16_kernel<<<dim3((unsigned)(S / 8 / 256)), blk, 0, stream>>>(x1, x1b, S);
  f32_to_bf16_kernel<<<dim3((unsigned)(S / 8 / 256)), blk, 0, stream>>>(x2, x2b, S);
  convert_weight_T<<<dim3(D / 32, D / 8), blk, 0, stream>>>(Wq, WqT, D, D);
  convert_weight_T<<<dim3(D / 32, D / 8), blk, 0, stream>>>(Wk, WkT, D, D);
  convert_weight_T<<<dim3(D / 32, D / 8), blk, 0, stream>>>(Wv, WvT, D, D);
  convert_weight_T<<<dim3(D / 32, Hh / 8), blk, 0, stream>>>(W1,      W1aT, D, Hh);
  convert_weight_T<<<dim3(D / 32, Hh / 8), blk, 0, stream>>>(W1 + Dh, W1bT, D, Hh);

  dim3 gD((D + TILE_N - 1) / TILE_N, M / TILE_M);    // (6, 256)
  dim3 gH((Hh + TILE_N - 1) / TILE_N, M / TILE_M);   // (2, 256)

  // 2) projections (async-LDS double-buffered bf16 WMMA, f32 accumulate)
  gemm_bf16_wmma_async<<<gD, blk, 0, stream>>>(x1b, WqT, Q1, nullptr, M, D, D, 0);
  gemm_bf16_wmma_async<<<gD, blk, 0, stream>>>(x2b, WqT, Q2, nullptr, M, D, D, 0);
  gemm_bf16_wmma_async<<<gD, blk, 0, stream>>>(x1b, WkT, K1, nullptr, M, D, D, 0);
  gemm_bf16_wmma_async<<<gD, blk, 0, stream>>>(x2b, WkT, K2, nullptr, M, D, D, 0);
  gemm_bf16_wmma_async<<<gD, blk, 0, stream>>>(x1b, WvT, V1, nullptr, M, D, D, 0);
  gemm_bf16_wmma_async<<<gD, blk, 0, stream>>>(x2b, WvT, V2, nullptr, M, D, D, 0);

  // 3) gate MLP first layer: H1 = x1@W1a + x2@W1b + b1 ; H2 = x2@W1a + x1@W1b + b1
  gemm_bf16_wmma_async<<<gH, blk, 0, stream>>>(x1b, W1aT, H1, nullptr, M, D, Hh, 0);
  gemm_bf16_wmma_async<<<gH, blk, 0, stream>>>(x2b, W1bT, H1, b1,      M, D, Hh, 1);
  gemm_bf16_wmma_async<<<gH, blk, 0, stream>>>(x2b, W1aT, H2, nullptr, M, D, Hh, 0);
  gemm_bf16_wmma_async<<<gH, blk, 0, stream>>>(x1b, W1bT, H2, b1,      M, D, Hh, 1);

  // 4) nk = noise @ Wk  (k_self = k + nk, since noise broadcasts over rows)
  noise_k_kernel<<<dim3((D + 255) / 256), blk, 0, stream>>>(noise, Wk, NK, D);

  // 5) fused epilogue: gate, 2-key softmax, residual output
  fuse_epilogue<<<dim3(M / 8), blk, 0, stream>>>(x1, Q1, K1, NK, K2, V1, V2, H1, W2, b2, out,     M);
  fuse_epilogue<<<dim3(M / 8), blk, 0, stream>>>(x2, Q2, K2, NK, K1, V2, V1, H2, W2, b2, out + S, M);
}
